// IDKN_Attention_10531259809803
// MI455X (gfx1250) — compile-verified, matrix operand order checked by reading, not run on hardware
//
#include <hip/hip_runtime.h>
#include <math.h>

// CDNA5 / gfx1250 implementation of the IDKN-Attention GAT pipeline.
// - v_wmma_f32_16x16x32_f16 for the dense feature transforms (3 shapes, templated)
// - Tensor Data Mover (tensor_load_to_lds + s_wait_tensorcnt) to stage W into LDS
// - B tile repacked in LDS to padded column-major f16 -> 2x ds_load_b128 per lane
// - sparse scatter/atomic kernels for edge softmax-aggregation and A@x4

typedef __attribute__((ext_vector_type(16))) _Float16 v16h;
typedef __attribute__((ext_vector_type(8)))  _Float16 v8h;
typedef __attribute__((ext_vector_type(8)))  float    v8f;
typedef unsigned int u32x4 __attribute__((ext_vector_type(4)));
typedef int          i32x4 __attribute__((ext_vector_type(4)));
typedef int          i32x8 __attribute__((ext_vector_type(8)));

#if defined(__has_builtin)
#if __has_builtin(__builtin_amdgcn_tensor_load_to_lds)
#define HAVE_TDM 1
#endif
#endif
#ifndef HAVE_TDM
#define HAVE_TDM 0
#endif

__device__ __forceinline__ float lrelu02(float x) { return x > 0.f ? x : 0.2f * x; }

__device__ __forceinline__ void atomicMaxF(float* addr, float val) {
    int* ia = (int*)addr;
    int cur = *ia;
    while (__int_as_float(cur) < val) {
        int prev = atomicCAS(ia, cur, __float_as_int(val));
        if (prev == cur) break;
        cur = prev;
    }
}

__device__ __forceinline__ void edge_sd(const int* __restrict__ ei, int E, int e,
                                        int& src, int& dst) {
    if (e < E) { src = ei[e]; dst = ei[E + e]; }
    else       { src = e - E; dst = e - E; }          // appended self-loops
}

// ---------------------------------------------------------------------------
// Fusion attention: q = x1@qw+qb, k = x2@kw+kb, v = [x1,x2]@vw+vb
// ---------------------------------------------------------------------------
__global__ void fuse_qkv_kernel(const float* __restrict__ x1, const float* __restrict__ x2,
                                const float* __restrict__ qw, const float* __restrict__ qb,
                                const float* __restrict__ kw, const float* __restrict__ kb,
                                const float* __restrict__ vw, const float* __restrict__ vb,
                                float* __restrict__ s, float* __restrict__ vout, int n) {
    int i = blockIdx.x * blockDim.x + threadIdx.x;
    if (i >= n) return;
    float a1[10], a2[10];
#pragma unroll
    for (int d = 0; d < 10; ++d) { a1[d] = x1[i * 10 + d]; a2[d] = x2[i * 10 + d]; }
    float acc = 0.f;
#pragma unroll
    for (int c = 0; c < 16; ++c) {
        float q = qb[c], k = kb[c], v = vb[c];
#pragma unroll
        for (int d = 0; d < 10; ++d) {
            q += a1[d] * qw[d * 16 + c];
            k += a2[d] * kw[d * 16 + c];
            v += a1[d] * vw[d * 16 + c] + a2[d] * vw[(10 + d) * 16 + c];
        }
        acc += q * k;
        vout[i * 16 + c] = v;
    }
    s[i] = acc;
}

// Single-block softmax reduction: red[0]=max, red[1]=sum(exp(s-max))
__global__ void softmax_reduce_kernel(const float* __restrict__ s, float* __restrict__ red, int n) {
    __shared__ float sm[1024];
    int t = threadIdx.x;
    float mx = -INFINITY;
    for (int i = t; i < n; i += blockDim.x) mx = fmaxf(mx, s[i]);
    sm[t] = mx; __syncthreads();
    for (int o = blockDim.x >> 1; o > 0; o >>= 1) {
        if (t < o) sm[t] = fmaxf(sm[t], sm[t + o]);
        __syncthreads();
    }
    float gmax = sm[0]; __syncthreads();
    float sum = 0.f;
    for (int i = t; i < n; i += blockDim.x) sum += __expf(s[i] - gmax);
    sm[t] = sum; __syncthreads();
    for (int o = blockDim.x >> 1; o > 0; o >>= 1) {
        if (t < o) sm[t] += sm[t + o];
        __syncthreads();
    }
    if (t == 0) { red[0] = gmax; red[1] = sm[0]; }
}

__global__ void fuse_scale_kernel(const float* __restrict__ s, const float* __restrict__ red,
                                  const float* __restrict__ v, float* __restrict__ fused, int n) {
    int i = blockIdx.x * blockDim.x + threadIdx.x;
    if (i >= n) return;
    float att = __expf(s[i] - red[0]) / red[1];
#pragma unroll
    for (int c = 0; c < 16; ++c) fused[i * 16 + c] = att * v[i * 16 + c];
}

// ---------------------------------------------------------------------------
// WMMA GEMM: Y[n, NOUT] = X[n, KIN] @ W[KIN, NOUT]; KIN in {16,32}, NOUT in {8,16,32}
// 256-thread block = 8 waves; wave w computes M-tile (blockIdx.x*8 + w).
// Stage raw fp32 W into LDS once per block (TDM), repack to a zero-padded
// column-major f16 tile sB[n][k] so each lane's B fetch is 2x ds_load_b128,
// then ONE v_wmma_f32_16x16x32_f16 per wave (K<=32 fits a single instruction).
// Per-lane layouts (ISA 7.12.2):
//   A (16x32 f16):  row M = lane&15;  k(j) = (j&7) + 8*(lane>=16) + 16*(j>=8)
//   B (32x16 f16):  col N = lane&15;  same k(j) mapping
//   C/D (16x16 f32): VGPR r -> row M = r + 8*(lane>=16), col N = lane&15
// ---------------------------------------------------------------------------
template <int KIN, int NOUT>
__global__ __launch_bounds__(256) void gemm_wmma_kernel(const float* __restrict__ X,
                                                        const float* __restrict__ W,
                                                        float* __restrict__ Y) {
    __shared__ __align__(16) float    sW[KIN * NOUT]; // raw fp32 weight tile (TDM dest)
    __shared__ __align__(16) _Float16 sB[16 * 32];    // padded col-major f16 [n][k]
    const int tid  = threadIdx.x;
    const int wave = tid >> 5;
    const int lane = tid & 31;
    const int hi   = lane >> 4;                       // 0: lanes 0-15, 1: lanes 16-31
    const int l    = lane & 15;
    const int tm   = (blockIdx.x * 8 + wave) * 16;
    const int tn   = (NOUT > 16) ? (int)blockIdx.y * 16 : 0;   // compile-time 0 otherwise

#if HAVE_TDM
    if (wave == 0) {
        // Tensor DMA Descriptor (ISA ch.8): 2D fp32 tensor, tile == whole tensor.
        const unsigned long long ga = (unsigned long long)W;
        const unsigned lds_base = (unsigned)(unsigned long long)(void*)&sW[0];
        // Group 0: [1:0]=count=1 | [63:32]=lds_addr | [120:64]=global_addr | [127:126]=type=2
        u32x4 g0 = { 1u,
                     lds_base,
                     (unsigned)ga,
                     (unsigned)((ga >> 32) & 0x01FFFFFFull) | (2u << 30) };
        // Group 1: [17:16]=data_size(4B)=2 | dim0=NOUT | dim1=KIN | tile0=NOUT |
        //          tile1=KIN | tile2=0 | dim0_stride=NOUT | dim1_stride=0
        i32x8 g1 = { (int)(2u << 16),
                     (int)(((unsigned)NOUT & 0xFFFFu) << 16),
                     (int)((((unsigned)NOUT >> 16) & 0xFFFFu) | (((unsigned)KIN & 0xFFFFu) << 16)),
                     (int)((((unsigned)KIN >> 16) & 0xFFFFu) | (((unsigned)NOUT & 0xFFFFu) << 16)),
                     (int)((unsigned)KIN & 0xFFFFu),
                     (int)NOUT,
                     0, 0 };
        i32x4 gz = { 0, 0, 0, 0 };
#if __clang_major__ >= 23
        i32x8 gz8 = { 0, 0, 0, 0, 0, 0, 0, 0 };
        __builtin_amdgcn_tensor_load_to_lds(g0, g1, gz, gz, gz8, 0);
#else
        __builtin_amdgcn_tensor_load_to_lds(g0, g1, gz, gz, 0);
#endif
#if __has_builtin(__builtin_amdgcn_s_wait_tensorcnt)
        __builtin_amdgcn_s_wait_tensorcnt(0);
#endif
    }
#else
    for (int idx = tid; idx < KIN * NOUT; idx += 256) sW[idx] = W[idx];
#endif
    __syncthreads();

    // Repack into zero-padded column-major f16: sB[n*32+k] = W[k, tn+n] (bounds -> 0)
#pragma unroll
    for (int idx = tid; idx < 16 * 32; idx += 256) {
        const int n = idx >> 5, k = idx & 31;
        float v = 0.f;
        if (k < KIN && (tn + n) < NOUT) v = sW[k * NOUT + tn + n];
        sB[n * 32 + k] = (_Float16)v;
    }
    __syncthreads();

    // A tile: row tm+l; k-runs {hi*8 .. +7} and {hi*8+16 .. +23} -> contiguous b128 loads
    float av[16];
    const float* xr = X + (size_t)(tm + l) * KIN + hi * 8;
    *(float4*)&av[0] = *(const float4*)(xr);
    *(float4*)&av[4] = *(const float4*)(xr + 4);
    if (KIN == 32) {
        *(float4*)&av[8]  = *(const float4*)(xr + 16);
        *(float4*)&av[12] = *(const float4*)(xr + 20);
    } else {
#pragma unroll
        for (int j = 8; j < 16; ++j) av[j] = 0.f;     // K padded to 32 with zeros
    }

    v16h a, b;
#pragma unroll
    for (int j = 0; j < 16; ++j) a[j] = (_Float16)av[j];

    // B tile: col l; same k-runs, contiguous in sB -> 2x ds_load_b128
    const _Float16* bp = &sB[l * 32 + hi * 8];
    const v8h b0 = *(const v8h*)(bp);
    const v8h b1 = *(const v8h*)(bp + 16);
#pragma unroll
    for (int j = 0; j < 8; ++j) { b[j] = b0[j]; b[8 + j] = b1[j]; }

    v8f c = {};
    c = __builtin_amdgcn_wmma_f32_16x16x32_f16(
        /*neg_a=*/false, a, /*neg_b=*/false, b,
        /*c_mod=*/(short)0, c, /*reuse_a=*/false, /*reuse_b=*/false);

    constexpr bool fullN = (NOUT % 16) == 0;          // grid exactly tiles columns
#pragma unroll
    for (int r = 0; r < 8; ++r) {
        if (fullN || (tn + l) < NOUT)
            Y[(size_t)(tm + r + hi * 8) * NOUT + tn + l] = c[r];
    }
}

// ---------------------------------------------------------------------------
// GAT per-node prep: attention coefficients as_/ad_, init m=-inf, den=0, out=bias
// ---------------------------------------------------------------------------
__global__ void gat_node_prep_kernel(const float* __restrict__ h,
                                     const float* __restrict__ a_src, const float* __restrict__ a_dst,
                                     const float* __restrict__ bias,
                                     float* __restrict__ as_, float* __restrict__ ad_,
                                     float* __restrict__ m, float* __restrict__ den,
                                     float* __restrict__ out, int n, int H, int C) {
    int i = blockIdx.x * blockDim.x + threadIdx.x;
    if (i >= n) return;
    const int HC = H * C;
    for (int hh = 0; hh < H; ++hh) {
        float sa = 0.f, sd = 0.f;
        for (int c = 0; c < C; ++c) {
            float hv = h[i * HC + hh * C + c];
            sa += hv * a_src[hh * C + c];
            sd += hv * a_dst[hh * C + c];
        }
        as_[i * H + hh] = sa;
        ad_[i * H + hh] = sd;
        m[i * H + hh]   = -INFINITY;
        den[i * H + hh] = 0.f;
    }
    for (int c = 0; c < HC; ++c) out[i * HC + c] = bias[c];
}

// Pass 1: segment max of edge logits into m[dst,h]
__global__ void gat_edge_max_kernel(const int* __restrict__ ei, int E, int n,
                                    const float* __restrict__ as_, const float* __restrict__ ad_,
                                    float* __restrict__ m, int H) {
    int e = blockIdx.x * blockDim.x + threadIdx.x;
    if (e >= E + n) return;
    int src, dst; edge_sd(ei, E, e, src, dst);
    for (int hh = 0; hh < H; ++hh) {
        float ev = lrelu02(as_[src * H + hh] + ad_[dst * H + hh]);
        atomicMaxF(&m[dst * H + hh], ev);
    }
}

// Pass 2: den[dst,h] += exp(logit - m[dst,h]) (logit recomputed: cheaper than 8.6MB buffer)
__global__ void gat_edge_sum_kernel(const int* __restrict__ ei, int E, int n,
                                    const float* __restrict__ as_, const float* __restrict__ ad_,
                                    const float* __restrict__ m, float* __restrict__ den, int H) {
    int e = blockIdx.x * blockDim.x + threadIdx.x;
    if (e >= E + n) return;
    int src, dst; edge_sd(ei, E, e, src, dst);
    for (int hh = 0; hh < H; ++hh) {
        float ev = lrelu02(as_[src * H + hh] + ad_[dst * H + hh]);
        atomicAdd(&den[dst * H + hh], __expf(ev - m[dst * H + hh]));
    }
}

// Pass 3: out[dst] += alpha * h[src]
__global__ void gat_edge_aggr_kernel(const int* __restrict__ ei, int E, int n,
                                     const float* __restrict__ as_, const float* __restrict__ ad_,
                                     const float* __restrict__ m, const float* __restrict__ den,
                                     const float* __restrict__ h, float* __restrict__ out,
                                     int H, int C) {
    int e = blockIdx.x * blockDim.x + threadIdx.x;
    if (e >= E + n) return;
    int src, dst; edge_sd(ei, E, e, src, dst);
    const int HC = H * C;
    for (int hh = 0; hh < H; ++hh) {
        float ev    = lrelu02(as_[src * H + hh] + ad_[dst * H + hh]);
        float alpha = __expf(ev - m[dst * H + hh]) / (den[dst * H + hh] + 1e-16f);
        for (int c = 0; c < C; ++c)
            atomicAdd(&out[dst * HC + hh * C + c], alpha * h[src * HC + hh * C + c]);
    }
}

// ---------------------------------------------------------------------------
// Sparse A@x4 (A = row-normalized adjacency with unit diagonal).
// Dense A would be 268MB of traffic; sparse form is ~10MB.
// ---------------------------------------------------------------------------
__global__ void adj_init_kernel(const float* __restrict__ x4, float* __restrict__ deg,
                                float* __restrict__ R3, int n) {
    int i = blockIdx.x * blockDim.x + threadIdx.x;
    if (i >= n) return;
    deg[i] = 1.0f;                               // unit diagonal
#pragma unroll
    for (int c = 0; c < 8; ++c) R3[i * 8 + c] = x4[i * 8 + c];
}

__global__ void adj_edges_kernel(const int* __restrict__ ei, int E,
                                 const float* __restrict__ x4,
                                 float* __restrict__ deg, float* __restrict__ R3) {
    int e = blockIdx.x * blockDim.x + threadIdx.x;
    if (e >= E) return;
    int src = ei[e], dst = ei[E + e];
    atomicAdd(&deg[src], 1.0f);
#pragma unroll
    for (int c = 0; c < 8; ++c) atomicAdd(&R3[src * 8 + c], x4[dst * 8 + c]);
}

__global__ void final_score_kernel(const float* __restrict__ x4, const float* __restrict__ deg,
                                   const float* __restrict__ R3, const float* __restrict__ lw,
                                   float* __restrict__ out, int n) {
    int i = blockIdx.x * blockDim.x + threadIdx.x;
    if (i >= n) return;
    float inv = 1.0f / fmaxf(deg[i], 1e-12f);
    float loc = 0.f, glob = 0.f;
#pragma unroll
    for (int c = 0; c < 8; ++c) {
        float r = R3[i * 8 + c] * inv;
        loc  += x4[i * 8 + c] * r;
        glob += r * lw[c];
    }
    out[i] = loc + glob;
}

// ---------------------------------------------------------------------------
extern "C" void kernel_launch(void* const* d_in, const int* in_sizes, int n_in,
                              void* d_out, int out_size, void* d_ws, size_t ws_size,
                              hipStream_t stream) {
    const float* x1   = (const float*)d_in[0];
    const float* x2   = (const float*)d_in[1];
    const int*   ei   = (const int*)  d_in[2];
    // d_in[3] = num_nodes (device scalar; use in_sizes instead)
    const float* q_w  = (const float*)d_in[4];
    const float* q_b  = (const float*)d_in[5];
    const float* k_w  = (const float*)d_in[6];
    const float* k_b  = (const float*)d_in[7];
    const float* v_w  = (const float*)d_in[8];
    const float* v_b  = (const float*)d_in[9];
    const float* W1   = (const float*)d_in[10];
    const float* a1s  = (const float*)d_in[11];
    const float* a1d  = (const float*)d_in[12];
    const float* b1   = (const float*)d_in[13];
    const float* W2   = (const float*)d_in[14];
    const float* a2s  = (const float*)d_in[15];
    const float* a2d  = (const float*)d_in[16];
    const float* b2   = (const float*)d_in[17];
    const float* W3   = (const float*)d_in[18];
    const float* a3s  = (const float*)d_in[19];
    const float* a3d  = (const float*)d_in[20];
    const float* b3   = (const float*)d_in[21];
    const float* lw   = (const float*)d_in[22];
    float* out = (float*)d_out;

    const int N = in_sizes[0] / 10;      // 8192
    const int E = in_sizes[2] / 2;       // 262144
    const int Etot = E + N;

    // ---- workspace carve-up (256B aligned) ----
    char* base = (char*)d_ws;
    size_t off = 0;
    auto alloc = [&](size_t floats) -> float* {
        float* p = (float*)(base + off);
        off += ((floats * sizeof(float)) + 255) & ~(size_t)255;
        return p;
    };
    float* s_buf = alloc(N);
    float* red   = alloc(2);
    float* v_buf = alloc((size_t)N * 16);
    float* fused = alloc((size_t)N * 16);
    float* h_buf = alloc((size_t)N * 32);
    float* as_b  = alloc((size_t)N * 8);
    float* ad_b  = alloc((size_t)N * 8);
    float* m_b   = alloc((size_t)N * 8);
    float* den_b = alloc((size_t)N * 8);
    float* h1    = alloc((size_t)N * 32);
    float* h2    = alloc((size_t)N * 16);
    float* x4    = alloc((size_t)N * 8);
    float* deg   = alloc(N);
    float* R3    = alloc((size_t)N * 8);
    (void)ws_size;

    const int TB = 256;
    const int nBlkN  = (N + TB - 1) / TB;
    const int nBlkE  = (Etot + TB - 1) / TB;
    const int nBlkEo = (E + TB - 1) / TB;
    const int gemmBlkM = N / 128;        // 8 M-tiles (8 waves) per block

    // ---- fusion attention ----
    fuse_qkv_kernel<<<nBlkN, TB, 0, stream>>>(x1, x2, q_w, q_b, k_w, k_b, v_w, v_b,
                                              s_buf, v_buf, N);
    softmax_reduce_kernel<<<1, 1024, 0, stream>>>(s_buf, red, N);
    fuse_scale_kernel<<<nBlkN, TB, 0, stream>>>(s_buf, red, v_buf, fused, N);

    // ---- GAT non-GEMM passes ----
    auto gat_rest = [&](const float* asW, const float* adW, const float* bias,
                        int H, int C, float* OUT) {
        gat_node_prep_kernel<<<nBlkN, TB, 0, stream>>>(h_buf, asW, adW, bias,
                                                       as_b, ad_b, m_b, den_b, OUT, N, H, C);
        gat_edge_max_kernel<<<nBlkE, TB, 0, stream>>>(ei, E, N, as_b, ad_b, m_b, H);
        gat_edge_sum_kernel<<<nBlkE, TB, 0, stream>>>(ei, E, N, as_b, ad_b, m_b, den_b, H);
        gat_edge_aggr_kernel<<<nBlkE, TB, 0, stream>>>(ei, E, N, as_b, ad_b, m_b, den_b,
                                                       h_buf, OUT, H, C);
    };

    // layer 1: GATConv(16 -> 8 heads x 4)
    gemm_wmma_kernel<16, 32><<<dim3(gemmBlkM, 2), 256, 0, stream>>>(fused, W1, h_buf);
    gat_rest(a1s, a1d, b1, 8, 4, h1);
    // layer 2: GATConv(32 -> 4 heads x 4)
    gemm_wmma_kernel<32, 16><<<dim3(gemmBlkM, 1), 256, 0, stream>>>(h1, W2, h_buf);
    gat_rest(a2s, a2d, b2, 4, 4, h2);
    // layer 3: GATConv(16 -> 2 heads x 4)
    gemm_wmma_kernel<16, 8><<<dim3(gemmBlkM, 1), 256, 0, stream>>>(h2, W3, h_buf);
    gat_rest(a3s, a3d, b3, 2, 4, x4);

    // ---- sparse A@x4 + scores ----
    adj_init_kernel<<<nBlkN, TB, 0, stream>>>(x4, deg, R3, N);
    adj_edges_kernel<<<nBlkEo, TB, 0, stream>>>(ei, E, x4, deg, R3);
    final_score_kernel<<<nBlkN, TB, 0, stream>>>(x4, deg, R3, lw, out, N);
}
